// LanguageModel_89472758710274
// MI455X (gfx1250) — matrix operand order repaired and checked by reading, not compiled, not run on hardware
//
#include <hip/hip_runtime.h>
#include <cmath>

// ---------------- model dims ----------------
#define KL 12
#define KH 768
#define KNH 12
#define KDH 64
#define KFF 3072
#define KV 21128
#define KB 64
#define KS 120
#define KTITLE 20
#define KT (KB * KS)   // 7680 tokens
#define KSP 128        // padded sequence for attention tiles

// ---------------- WMMA types ----------------
typedef __attribute__((ext_vector_type(16))) __bf16 v16bf;
typedef __attribute__((ext_vector_type(8))) float v8f;

__device__ inline unsigned short f2bf(float f) {
  unsigned u = __builtin_bit_cast(unsigned, f);
  unsigned r = u + 0x7FFFu + ((u >> 16) & 1u);
  return (unsigned short)(r >> 16);
}

__device__ inline v8f wmma_bf16(v16bf a, v16bf b, v8f c) {
  // D = A(16x32 bf16) x B(32x16 bf16) + C(16x16 f32)
  return __builtin_amdgcn_wmma_f32_16x16x32_bf16(false, a, false, b, (short)0, c,
                                                 false, false);
}

// A fragment: A is 16x32 row-major in LDS at (row0, k0), row stride ld (ushorts).
// lanes 0-15: M=lane, K=k0+{0..7,16..23}; lanes 16-31: M=lane-16, K=k0+{8..15,24..31}
__device__ inline v16bf load_a_frag(const unsigned short* sm, int row0, int k0, int ld) {
  int lane = threadIdx.x & 31;
  const unsigned short* p = sm + (size_t)(row0 + (lane & 15)) * ld + k0 + ((lane >> 4) << 3);
  union { uint4 q[2]; v16bf v; } c;
  c.q[0] = *(const uint4*)p;
  c.q[1] = *(const uint4*)(p + 16);
  return c.v;
}

// B fragment from TRANSPOSED storage: LDS holds Bt[n][k] row-major.
// lanes 0-15: N=col0+lane, K=k0+{0..7,16..23}; lanes 16-31: N=col0+lane-16, K=k0+{8..15,24..31}
__device__ inline v16bf load_bT_frag(const unsigned short* sm, int col0, int k0, int ld) {
  int lane = threadIdx.x & 31;
  const unsigned short* p = sm + (size_t)(col0 + (lane & 15)) * ld + k0 + ((lane >> 4) << 3);
  union { uint4 q[2]; v16bf v; } c;
  c.q[0] = *(const uint4*)p;
  c.q[1] = *(const uint4*)(p + 16);
  return c.v;
}

__device__ inline float block_reduce_sum(float v, float* sbuf /* >= 9 floats */) {
  __syncthreads();
  int lane = threadIdx.x & 31, wid = threadIdx.x >> 5;
#pragma unroll
  for (int m = 16; m >= 1; m >>= 1) v += __shfl_xor(v, m, 32);
  if (lane == 0) sbuf[wid] = v;
  __syncthreads();
  if (threadIdx.x == 0) {
    float s = 0.f;
#pragma unroll
    for (int i = 0; i < 8; ++i) s += sbuf[i];
    sbuf[8] = s;
  }
  __syncthreads();
  return sbuf[8];
}

// ---------------- fp32 -> bf16 transpose conversion: out[n][k] = bf16(in[k][n]) ----------------
__global__ __launch_bounds__(256) void cvt_transpose_kernel(
    const float* __restrict__ in, unsigned short* __restrict__ out, int K, int N) {
  __shared__ unsigned short tile[32][33];
  size_t base = (size_t)blockIdx.z * (size_t)K * N;
  int k0 = blockIdx.y * 32, n0 = blockIdx.x * 32;
  int tx = threadIdx.x & 31, ty = threadIdx.x >> 5;  // 8 row-groups
#pragma unroll
  for (int r = ty; r < 32; r += 8) {
    int k = k0 + r, n = n0 + tx;
    unsigned short v = 0;
    if (k < K && n < N) v = f2bf(in[base + (size_t)k * N + n]);
    tile[r][tx] = v;
  }
  __syncthreads();
#pragma unroll
  for (int r = ty; r < 32; r += 8) {
    int n = n0 + r, k = k0 + tx;
    if (n < N && k < K) out[base + (size_t)n * K + k] = tile[tx][r];
  }
}

// ---------------- embedding + LayerNorm ----------------
__global__ __launch_bounds__(256) void embed_ln_kernel(
    const int* __restrict__ x, const float* __restrict__ we, const float* __restrict__ pe,
    const float* __restrict__ te, const float* __restrict__ lw, const float* __restrict__ lb,
    float* __restrict__ hF, unsigned short* __restrict__ hB) {
  __shared__ float sbuf[9];
  int t = blockIdx.x;
  int s = t % KS;
  int tok = x[t];
  float v[3];
  float s1 = 0.f;
#pragma unroll
  for (int i = 0; i < 3; ++i) {
    int idx = threadIdx.x + i * 256;
    float val = we[(size_t)tok * KH + idx] + pe[(size_t)s * KH + idx] + te[idx];
    v[i] = val;
    s1 += val;
  }
  float mean = block_reduce_sum(s1, sbuf) * (1.0f / KH);
  float s2 = 0.f;
#pragma unroll
  for (int i = 0; i < 3; ++i) { float d = v[i] - mean; s2 += d * d; }
  float var = block_reduce_sum(s2, sbuf) * (1.0f / KH);
  float inv = rsqrtf(var + 1e-12f);
#pragma unroll
  for (int i = 0; i < 3; ++i) {
    int idx = threadIdx.x + i * 256;
    float o = (v[i] - mean) * inv * lw[idx] + lb[idx];
    hF[(size_t)t * KH + idx] = o;
    hB[(size_t)t * KH + idx] = f2bf(o);
  }
}

// ---------------- residual add + LayerNorm ----------------
__global__ __launch_bounds__(256) void add_ln_kernel(
    const float* __restrict__ res, const float* __restrict__ delta,
    const float* __restrict__ lw, const float* __restrict__ lb,
    float* __restrict__ outF, unsigned short* __restrict__ outB) {
  __shared__ float sbuf[9];
  int t = blockIdx.x;
  float v[3];
  float s1 = 0.f;
#pragma unroll
  for (int i = 0; i < 3; ++i) {
    int idx = threadIdx.x + i * 256;
    float val = res[(size_t)t * KH + idx] + delta[(size_t)t * KH + idx];
    v[i] = val;
    s1 += val;
  }
  float mean = block_reduce_sum(s1, sbuf) * (1.0f / KH);
  float s2 = 0.f;
#pragma unroll
  for (int i = 0; i < 3; ++i) { float d = v[i] - mean; s2 += d * d; }
  float var = block_reduce_sum(s2, sbuf) * (1.0f / KH);
  float inv = rsqrtf(var + 1e-12f);
#pragma unroll
  for (int i = 0; i < 3; ++i) {
    int idx = threadIdx.x + i * 256;
    float o = (v[i] - mean) * inv * lw[idx] + lb[idx];
    outF[(size_t)t * KH + idx] = o;
    outB[(size_t)t * KH + idx] = f2bf(o);
  }
}

// ---------------- generic WMMA GEMM:  C[M,N] = act(A[M,K] @ W[K,N] + bias) ----------------
// 128x128 block tile, 8 waves, wave = 32x64 strip (acc[2][4]).
// Staging pinned via sched_group_barrier: 8 VMEM reads, then 8 DS writes
// -> staggered s_wait_loadcnt, one latency per tile.
// A bf16 row-major [M,K]; WT bf16 = W^T row-major [N,K].
// M%128==0, N%128==0, K%64==0. act: 0=identity, 1=exact GELU.
// vt==1: write bf16 output in per-head transposed V layout [b,h,d][s] (stride KSP).
__global__ __launch_bounds__(256, 2) void wmma_gemm_kernel(
    const unsigned short* __restrict__ A, const unsigned short* __restrict__ WT,
    const float* __restrict__ bias, float* __restrict__ outF,
    unsigned short* __restrict__ outB, int M, int N, int K, int act, int vt) {
  __shared__ __align__(16) unsigned short sA[128][72];
  __shared__ __align__(16) unsigned short sBT[128][72];
  int tid = threadIdx.x;
  int wid = tid >> 5, lane = tid & 31;
  int bm = blockIdx.y * 128;
  int bn = blockIdx.x * 128;
  int mt = wid >> 1;  // 0..3 -> 32-row strip
  int nh = wid & 1;   // 0..1 -> 64-col strip
  v8f acc[2][4] = {};

  // staging: 128 rows x 64 cols per array; 2 threads/row, 4 uint4 each
  int rs = tid >> 1, csb = (tid & 1) << 5;
  const unsigned short* gA = A + (size_t)(bm + rs) * K + csb;
  const unsigned short* gB = WT + (size_t)(bn + rs) * K + csb;

  for (int k0 = 0; k0 < K; k0 += 64) {
    uint4 ta[4], tb[4];
#pragma unroll
    for (int j = 0; j < 4; ++j) ta[j] = *(const uint4*)(gA + k0 + j * 8);
#pragma unroll
    for (int j = 0; j < 4; ++j) tb[j] = *(const uint4*)(gB + k0 + j * 8);
#pragma unroll
    for (int j = 0; j < 4; ++j) *(uint4*)&sA[rs][csb + j * 8] = ta[j];
#pragma unroll
    for (int j = 0; j < 4; ++j) *(uint4*)&sBT[rs][csb + j * 8] = tb[j];
    // pin the schedule of this region: all 8 global loads, then all 8 LDS stores
    __builtin_amdgcn_sched_group_barrier(0x020, 8, 0);  // VMEM read x8
    __builtin_amdgcn_sched_group_barrier(0x200, 8, 0);  // DS write x8
    __syncthreads();
#pragma unroll
    for (int kk = 0; kk < 64; kk += 32) {
      v16bf a0 = load_a_frag(&sA[0][0], mt * 32, kk, 72);
      v16bf a1 = load_a_frag(&sA[0][0], mt * 32 + 16, kk, 72);
#pragma unroll
      for (int t = 0; t < 4; ++t) {
        v16bf bt = load_bT_frag(&sBT[0][0], nh * 64 + t * 16, kk, 72);
        acc[0][t] = wmma_bf16(a0, bt, acc[0][t]);
        acc[1][t] = wmma_bf16(a1, bt, acc[1][t]);
      }
    }
    __syncthreads();
  }

  int col = lane & 15, rbase = (lane >> 4) << 3;
#pragma unroll
  for (int sub = 0; sub < 2; ++sub) {
#pragma unroll
    for (int t = 0; t < 4; ++t) {
#pragma unroll
      for (int e = 0; e < 8; ++e) {
        int row = bm + mt * 32 + sub * 16 + rbase + e;
        int cg = bn + nh * 64 + t * 16 + col;
        float v = acc[sub][t][e] + bias[cg];
        if (act == 1) v = 0.5f * v * (1.0f + erff(v * 0.70710678f));
        if (outF) outF[(size_t)row * N + cg] = v;
        if (outB) {
          if (vt) {  // per-head transposed V: [b, h, d][s]
            int b = row / KS, s = row % KS;
            int hh = cg >> 6, d = cg & 63;
            outB[(((size_t)b * KNH + hh) * KDH + d) * KSP + s] = f2bf(v);
          } else {
            outB[(size_t)row * N + cg] = f2bf(v);
          }
        }
      }
    }
  }
}

// ---------------- attention: scores + masked softmax (per batch*head) ----------------
__global__ __launch_bounds__(256) void attn_scores_kernel(
    const unsigned short* __restrict__ qb, const unsigned short* __restrict__ kb,
    unsigned short* __restrict__ attn) {
  __shared__ __align__(16) unsigned short sQ[KSP][72];
  __shared__ __align__(16) unsigned short sK[KSP][72];
  int bh = blockIdx.x;
  int b = bh / KNH, hh = bh % KNH;
  int tid = threadIdx.x;
  for (int i = tid; i < KSP * KDH; i += 256) {
    int r = i >> 6, c = i & 63;
    unsigned short qv = 0, kv = 0;
    if (r < KS) {
      size_t off = ((size_t)(b * KS + r)) * KH + hh * KDH + c;
      qv = qb[off];
      kv = kb[off];
    }
    sQ[r][c] = qv;
    sK[r][c] = kv;
  }
  __syncthreads();

  int wid = tid >> 5, lane = tid & 31;
  int mrow = wid * 16;  // query strip
  v8f acc[8] = {};
#pragma unroll
  for (int k0 = 0; k0 < KDH; k0 += 32) {
    v16bf a = load_a_frag(&sQ[0][0], mrow, k0, 72);
#pragma unroll
    for (int nt = 0; nt < 8; ++nt) {
      v16bf bt = load_bT_frag(&sK[0][0], nt * 16, k0, 72);
      acc[nt] = wmma_bf16(a, bt, acc[nt]);
    }
  }

  int col = lane & 15, rbase = (lane >> 4) << 3;
  const float scale = 0.125f;  // 1/sqrt(64)
#pragma unroll
  for (int e = 0; e < 8; ++e) {
    int row = mrow + rbase + e;  // query index (may be padding >= KS)
    float vals[8];
    float m = -3.0e38f;
#pragma unroll
    for (int nt = 0; nt < 8; ++nt) {
      int kc = nt * 16 + col;
      float v = acc[nt][e] * scale;
      bool ok = (kc < KS) && ((row < KTITLE) ? (kc < KTITLE) : ((kc < KTITLE) || (kc <= row)));
      v = ok ? v : -3.0e38f;
      vals[nt] = v;
      m = fmaxf(m, v);
    }
#pragma unroll
    for (int msk = 1; msk < 16; msk <<= 1) m = fmaxf(m, __shfl_xor(m, msk, 32));
    float s = 0.f;
#pragma unroll
    for (int nt = 0; nt < 8; ++nt) {
      float p = __expf(vals[nt] - m);
      vals[nt] = p;
      s += p;
    }
#pragma unroll
    for (int msk = 1; msk < 16; msk <<= 1) s += __shfl_xor(s, msk, 32);
    float inv = 1.0f / s;
#pragma unroll
    for (int nt = 0; nt < 8; ++nt) {
      attn[((size_t)bh * KSP + row) * KSP + nt * 16 + col] = f2bf(vals[nt] * inv);
    }
  }
}

// ---------------- attention: ctx = P @ V (per batch*head), V pre-transposed ----------------
__global__ __launch_bounds__(256) void attn_ctx_kernel(
    const unsigned short* __restrict__ attn, const unsigned short* __restrict__ vtb,
    unsigned short* __restrict__ ctx) {
  __shared__ __align__(16) unsigned short sP[KSP][136];
  __shared__ __align__(16) unsigned short sVT[KDH][136];  // [dh][token]
  int bh = blockIdx.x;
  int b = bh / KNH, hh = bh % KNH;
  int tid = threadIdx.x;
  for (int i = tid; i < (KSP * KSP) / 8; i += 256) {
    int r = i >> 4, c8 = (i & 15) << 3;
    *(uint4*)&sP[r][c8] = *(const uint4*)(attn + ((size_t)bh * KSP + r) * KSP + c8);
  }
  for (int i = tid; i < KDH * KSP; i += 256) {
    int d = i >> 7, c = i & 127;
    unsigned short vv = 0;
    if (c < KS) vv = vtb[((size_t)bh * KDH + d) * KSP + c];
    sVT[d][c] = vv;
  }
  __syncthreads();

  int wid = tid >> 5, lane = tid & 31;
  int mrow = wid * 16;
  int col = lane & 15, rbase = (lane >> 4) << 3;
#pragma unroll
  for (int nt = 0; nt < 4; ++nt) {  // 64 dh cols
    v8f acc = {};
#pragma unroll
    for (int k0 = 0; k0 < KSP; k0 += 32) {
      v16bf a = load_a_frag(&sP[0][0], mrow, k0, 136);
      v16bf bf = load_bT_frag(&sVT[0][0], nt * 16, k0, 136);
      acc = wmma_bf16(a, bf, acc);
    }
#pragma unroll
    for (int e = 0; e < 8; ++e) {
      int row = mrow + rbase + e;
      if (row < KS) {
        ctx[((size_t)(b * KS + row)) * KH + hh * KDH + nt * 16 + col] = f2bf(acc[e]);
      }
    }
  }
}

// ---------------- classifier + online log-softmax NLL (no logits tensor) ----------------
// WT = cls_w transposed: [V][KH] bf16. BM=64 tokens, V chunks of 128 columns.
// Wave = 16 rows x 64 cols (acc[4]); sched_group_barrier-pinned staging.
__global__ __launch_bounds__(256, 2) void cls_nll_kernel(
    const unsigned short* __restrict__ Abf, const unsigned short* __restrict__ WT,
    const float* __restrict__ bias, const int* __restrict__ y, float* __restrict__ nll) {
  __shared__ __align__(16) unsigned short sA[64][72];
  __shared__ __align__(16) unsigned short sBT[128][72];
  __shared__ float wmax[64][2], wsum[64][2];
  __shared__ float rowm[64], rowl[64], rowt[64];
  __shared__ int ytok[64];
  int tid = threadIdx.x, wid = tid >> 5, lane = tid & 31;
  int bm = blockIdx.x * 64;
  if (tid < 64) {
    rowm[tid] = -3.0e38f;
    rowl[tid] = 0.f;
    rowt[tid] = 0.f;
    ytok[tid] = y[bm + tid];
  }
  __syncthreads();

  int mt = wid >> 1, nh = wid & 1;
  int col = lane & 15, rbase = (lane >> 4) << 3;
  // A staging: 64 rows x 64 cols, 4 threads/row, 2 uint4 each
  int ras = tid >> 2, cas = (tid & 3) << 3;
  const unsigned short* gA = Abf + (size_t)(bm + ras) * KH + cas;
  // B staging: 128 rows x 64 cols, 2 threads/row, 4 uint4 each
  int rbs = tid >> 1, cbs = (tid & 1) << 5;
  const int nchunks = (KV + 127) / 128;

  for (int ch = 0; ch < nchunks; ++ch) {
    int bn = ch * 128;
    bool brow_ok = (bn + rbs) < KV;
    const unsigned short* gB = WT + (size_t)(bn + rbs) * KH + cbs;
    v8f acc[4] = {};
    for (int k0 = 0; k0 < KH; k0 += 64) {
      uint4 tA0, tA1, tB[4];
      tA0 = *(const uint4*)(gA + k0);
      tA1 = *(const uint4*)(gA + k0 + 32);
      if (brow_ok) {
#pragma unroll
        for (int j = 0; j < 4; ++j) tB[j] = *(const uint4*)(gB + k0 + j * 8);
      } else {
        uint4 z = {0u, 0u, 0u, 0u};
#pragma unroll
        for (int j = 0; j < 4; ++j) tB[j] = z;
      }
      *(uint4*)&sA[ras][cas] = tA0;
      *(uint4*)&sA[ras][cas + 32] = tA1;
#pragma unroll
      for (int j = 0; j < 4; ++j) *(uint4*)&sBT[rbs][cbs + j * 8] = tB[j];
      __builtin_amdgcn_sched_group_barrier(0x020, 6, 0);  // VMEM read x6
      __builtin_amdgcn_sched_group_barrier(0x200, 6, 0);  // DS write x6
      __syncthreads();
#pragma unroll
      for (int kk = 0; kk < 64; kk += 32) {
        v16bf a = load_a_frag(&sA[0][0], mt * 16, kk, 72);
#pragma unroll
        for (int t = 0; t < 4; ++t) {
          v16bf bt = load_bT_frag(&sBT[0][0], nh * 64 + t * 16, kk, 72);
          acc[t] = wmma_bf16(a, bt, acc[t]);
        }
      }
      __syncthreads();
    }

    float vv[4][8];
#pragma unroll
    for (int e = 0; e < 8; ++e) {
      int rl = mt * 16 + rbase + e;
      float cm = -3.0e38f;
#pragma unroll
      for (int t = 0; t < 4; ++t) {
        int cg = bn + nh * 64 + t * 16 + col;
        float a0 = (cg < KV) ? acc[t][e] + bias[cg] : -3.0e38f;
        if (cg == ytok[rl]) rowt[rl] = a0;  // unique writer per row
        vv[t][e] = a0;
        cm = fmaxf(cm, a0);
      }
#pragma unroll
      for (int msk = 1; msk < 16; msk <<= 1) cm = fmaxf(cm, __shfl_xor(cm, msk, 32));
      if (col == 0) wmax[rl][nh] = cm;
    }
    __syncthreads();
    if (tid < 64) {
      float cm = fmaxf(wmax[tid][0], wmax[tid][1]);
      float mo = rowm[tid];
      float mn = fmaxf(mo, cm);
      rowl[tid] *= __expf(mo - mn);
      rowm[tid] = mn;
    }
    __syncthreads();
#pragma unroll
    for (int e = 0; e < 8; ++e) {
      int rl = mt * 16 + rbase + e;
      float mn = rowm[rl];
      float s = 0.f;
#pragma unroll
      for (int t = 0; t < 4; ++t) s += __expf(vv[t][e] - mn);
#pragma unroll
      for (int msk = 1; msk < 16; msk <<= 1) s += __shfl_xor(s, msk, 32);
      if (col == 0) wsum[rl][nh] = s;
    }
    __syncthreads();
    if (tid < 64) rowl[tid] += wsum[tid][0] + wsum[tid][1];
    __syncthreads();
  }
  if (tid < 64) nll[bm + tid] = (rowm[tid] + logf(rowl[tid])) - rowt[tid];
}

// ---------------- final mean ----------------
__global__ __launch_bounds__(256) void mean_kernel(const float* __restrict__ nll,
                                                   float* __restrict__ out) {
  __shared__ float sbuf[9];
  float s = 0.f;
  for (int i = threadIdx.x; i < KT; i += 256) s += nll[i];
  s = block_reduce_sum(s, sbuf);
  if (threadIdx.x == 0) out[0] = s / (float)KT;
}

// ---------------- host orchestration ----------------
static inline size_t align256(size_t x) { return (x + 255) & ~(size_t)255; }

extern "C" void kernel_launch(void* const* d_in, const int* in_sizes, int n_in,
                              void* d_out, int out_size, void* d_ws, size_t ws_size,
                              hipStream_t stream) {
  (void)in_sizes; (void)n_in; (void)out_size; (void)ws_size;
  const int* x = (const int*)d_in[0];
  const int* y = (const int*)d_in[1];
  const float* word_emb = (const float*)d_in[2];
  const float* pos_emb = (const float*)d_in[3];
  const float* type_emb = (const float*)d_in[4];
  const float* emb_ln_w = (const float*)d_in[5];
  const float* emb_ln_b = (const float*)d_in[6];
  const float* q_w = (const float*)d_in[7];
  const float* q_b = (const float*)d_in[8];
  const float* k_w = (const float*)d_in[9];
  const float* k_b = (const float*)d_in[10];
  const float* v_w = (const float*)d_in[11];
  const float* v_b = (const float*)d_in[12];
  const float* ao_w = (const float*)d_in[13];
  const float* ao_b = (const float*)d_in[14];
  const float* a_ln_w = (const float*)d_in[15];
  const float* a_ln_b = (const float*)d_in[16];
  const float* f1_w = (const float*)d_in[17];
  const float* f1_b = (const float*)d_in[18];
  const float* f2_w = (const float*)d_in[19];
  const float* f2_b = (const float*)d_in[20];
  const float* f_ln_w = (const float*)d_in[21];
  const float* f_ln_b = (const float*)d_in[22];
  const float* cls_w = (const float*)d_in[23];
  const float* cls_b = (const float*)d_in[24];
  float* out = (float*)d_out;

  const size_t HH = (size_t)KH * KH;
  const size_t HFF = (size_t)KH * KFF;
  const size_t W_ATT = (size_t)KL * HH;
  const size_t W_FFN = (size_t)KL * HFF;
  const size_t W_CLS = (size_t)KH * KV;
  const size_t TH = (size_t)KT * KH;
  const size_t TFF = (size_t)KT * KFF;
  const size_t ATT = (size_t)KB * KNH * KSP * KSP;
  const size_t VTSZ = (size_t)KB * KNH * KDH * KSP;

  char* p = (char*)d_ws;
  auto carve = [&](size_t bytes) { char* r = p; p += align256(bytes); return r; };

  unsigned short* wqT = (unsigned short*)carve(W_ATT * 2);
  unsigned short* wkT = (unsigned short*)carve(W_ATT * 2);
  unsigned short* wvT = (unsigned short*)carve(W_ATT * 2);
  unsigned short* waoT = (unsigned short*)carve(W_ATT * 2);
  unsigned short* wf1T = (unsigned short*)carve(W_FFN * 2);
  unsigned short* wf2T = (unsigned short*)carve(W_FFN * 2);
  unsigned short* wclsT = (unsigned short*)carve(W_CLS * 2);
  float* hA_f = (float*)carve(TH * 4);
  unsigned short* hA_b = (unsigned short*)carve(TH * 2);
  float* hB_f = (float*)carve(TH * 4);
  unsigned short* hB_b = (unsigned short*)carve(TH * 2);
  unsigned short* qact = (unsigned short*)carve(TH * 2);
  unsigned short* kact = (unsigned short*)carve(TH * 2);
  unsigned short* vtb = (unsigned short*)carve(VTSZ * 2);
  unsigned short* attnb = (unsigned short*)carve(ATT * 2);
  unsigned short* ctxb = (unsigned short*)carve(TH * 2);
  float* tmpf = (float*)carve(TH * 4);
  unsigned short* ffn1b = (unsigned short*)carve(TFF * 2);
  float* nllb = (float*)carve((size_t)KT * 4);

  // --- convert + transpose weights to bf16 W^T ---
  auto cvtT = [&](const float* s, unsigned short* d, int Kd, int Nd, int batch) {
    dim3 g((Nd + 31) / 32, (Kd + 31) / 32, batch);
    cvt_transpose_kernel<<<g, 256, 0, stream>>>(s, d, Kd, Nd);
  };
  cvtT(q_w, wqT, KH, KH, KL);
  cvtT(k_w, wkT, KH, KH, KL);
  cvtT(v_w, wvT, KH, KH, KL);
  cvtT(ao_w, waoT, KH, KH, KL);
  cvtT(f1_w, wf1T, KH, KFF, KL);
  cvtT(f2_w, wf2T, KFF, KH, KL);
  cvtT(cls_w, wclsT, KH, KV, 1);

  // --- embeddings + LN ---
  embed_ln_kernel<<<KT, 256, 0, stream>>>(x, word_emb, pos_emb, type_emb, emb_ln_w,
                                          emb_ln_b, hA_f, hA_b);

  auto gemm = [&](const unsigned short* A, const unsigned short* WT, const float* bias,
                  float* oF, unsigned short* oB, int M, int N, int K, int act, int vt) {
    dim3 g(N / 128, M / 128);
    wmma_gemm_kernel<<<g, 256, 0, stream>>>(A, WT, bias, oF, oB, M, N, K, act, vt);
  };

  for (int l = 0; l < KL; ++l) {
    const unsigned short* lwq = wqT + (size_t)l * HH;
    const unsigned short* lwk = wkT + (size_t)l * HH;
    const unsigned short* lwv = wvT + (size_t)l * HH;
    const unsigned short* lwao = waoT + (size_t)l * HH;
    const unsigned short* lwf1 = wf1T + (size_t)l * HFF;
    const unsigned short* lwf2 = wf2T + (size_t)l * HFF;

    gemm(hA_b, lwq, q_b + (size_t)l * KH, nullptr, qact, KT, KH, KH, 0, 0);
    gemm(hA_b, lwk, k_b + (size_t)l * KH, nullptr, kact, KT, KH, KH, 0, 0);
    gemm(hA_b, lwv, v_b + (size_t)l * KH, nullptr, vtb, KT, KH, KH, 0, 1);

    attn_scores_kernel<<<KB * KNH, 256, 0, stream>>>(qact, kact, attnb);
    attn_ctx_kernel<<<KB * KNH, 256, 0, stream>>>(attnb, vtb, ctxb);

    gemm(ctxb, lwao, ao_b + (size_t)l * KH, tmpf, nullptr, KT, KH, KH, 0, 0);
    add_ln_kernel<<<KT, 256, 0, stream>>>(hA_f, tmpf, a_ln_w + (size_t)l * KH,
                                          a_ln_b + (size_t)l * KH, hB_f, hB_b);

    gemm(hB_b, lwf1, f1_b + (size_t)l * KFF, nullptr, ffn1b, KT, KFF, KH, 1, 0);
    gemm(ffn1b, lwf2, f2_b + (size_t)l * KH, tmpf, nullptr, KT, KH, KFF, 0, 0);
    add_ln_kernel<<<KT, 256, 0, stream>>>(hB_f, tmpf, f_ln_w + (size_t)l * KH,
                                          f_ln_b + (size_t)l * KH, hA_f, hA_b);
  }

  // --- fused classifier + log-softmax + NLL ---
  cls_nll_kernel<<<KT / 64, 256, 0, stream>>>(hA_b, wclsT, cls_b, y, nllb);
  mean_kernel<<<1, 256, 0, stream>>>(nllb, out);
}
